// StreamingxLSTM_54485955117303
// MI455X (gfx1250) — matrix operand order repaired and checked by reading, not compiled, not run on hardware
//
#include <hip/hip_runtime.h>
#include <hip/hip_bf16.h>
#include <math.h>

// ---------------------------------------------------------------------------
// StreamingxLSTM step for MI455X (gfx1250, wave32, WMMA).
// GEMMs run in bf16 via v_wmma_f32_16x16x32_bf16; the C-state update is a
// single streaming pass fused with the C_t·q contraction (LDS ds_add_f32).
// ---------------------------------------------------------------------------

#define INP  768
#define H    8
#define D    96
#define HID  768
#define PROJ 1536
#define KER  4
#define FOUT 3088
#define CAP  15.0f

typedef __attribute__((ext_vector_type(16))) __bf16 v16bf;
typedef __attribute__((ext_vector_type(8)))  __bf16 v8bf;
typedef __attribute__((ext_vector_type(8)))  float  v8f;

// ------------------------------ fp32 -> bf16 -------------------------------
__global__ void f32_to_bf16_kernel(const float* __restrict__ in,
                                   __bf16* __restrict__ out, int n) {
    int i = blockIdx.x * blockDim.x + threadIdx.x;
    if (i < n) out[i] = (__bf16)in[i];
}

// ------------------------- LayerNorm -> bf16 -------------------------------
__global__ __launch_bounds__(256)
void ln_bf16_kernel(const float* __restrict__ x, const float* __restrict__ w,
                    const float* __restrict__ b, __bf16* __restrict__ out,
                    int dim) {
    const int row = blockIdx.x;
    const int tid = threadIdx.x;
    __shared__ float red[256];
    const float* xr = x + (size_t)row * dim;

    float s = 0.f, s2 = 0.f;
    for (int i = tid; i < dim; i += 256) { float v = xr[i]; s += v; s2 += v * v; }

    red[tid] = s; __syncthreads();
    for (int o = 128; o > 0; o >>= 1) { if (tid < o) red[tid] += red[tid + o]; __syncthreads(); }
    const float mean = red[0] / (float)dim;
    __syncthreads();
    red[tid] = s2; __syncthreads();
    for (int o = 128; o > 0; o >>= 1) { if (tid < o) red[tid] += red[tid + o]; __syncthreads(); }
    const float var  = red[0] / (float)dim - mean * mean;
    const float rstd = rsqrtf(var + 1e-6f);

    __bf16* orow = out + (size_t)row * dim;
    for (int i = tid; i < dim; i += 256) {
        float v = (xr[i] - mean) * rstd * w[i] + b[i];
        orow[i] = (__bf16)v;
    }
}

// --------------------- causal conv (K=4) + SiLU -> bf16 --------------------
__global__ void conv_silu_kernel(const float* __restrict__ xt,
                                 const float* __restrict__ cw,
                                 __bf16* __restrict__ out, int total) {
    int i = blockIdx.x * blockDim.x + threadIdx.x;
    if (i >= total) return;
    int b = i / PROJ, j = i - b * PROJ;
    const float* row = xt + (size_t)b * PROJ;
    float acc = 0.f;
#pragma unroll
    for (int k = 0; k < KER; ++k) {
        int idx = j + k - (KER - 1);
        if (idx >= 0) acc += cw[k] * row[idx];
    }
    float sig = 1.f / (1.f + expf(-acc));
    out[i] = (__bf16)(acc * sig);
}

// --------------------------- bf16 WMMA GEMM --------------------------------
// Out[M,N] = A[M,K] * W[N,K]^T (+ bias[N]) (+ addm[M,N])
// wave tile 64x32 (4x2 16x16 tiles), block 8 waves -> 128x128.
__global__ __launch_bounds__(256)
void wmma_gemm_bf16_kernel(const __bf16* __restrict__ A,
                           const __bf16* __restrict__ W,
                           float* __restrict__ Out,
                           const float* __restrict__ bias,
                           const float* __restrict__ addm,
                           int M, int N, int K) {
    const int lane = threadIdx.x & 31;
    const int wave = threadIdx.x >> 5;
    const int half = lane >> 4;   // 0/1 : K half-wave split
    const int l16  = lane & 15;

    const int m0 = blockIdx.y * 128 + (wave >> 2) * 64;
    const int n0 = blockIdx.x * 128 + (wave & 3) * 32;

    v8f acc[4][2];
#pragma unroll
    for (int i = 0; i < 4; ++i)
#pragma unroll
        for (int j = 0; j < 2; ++j) acc[i][j] = (v8f){0, 0, 0, 0, 0, 0, 0, 0};

    for (int k0 = 0; k0 < K; k0 += 32) {
        // A fragments (16x32 bf16): lane holds row m; K = half*16 .. +15 contig
        v16bf afrag[4];
#pragma unroll
        for (int mi = 0; mi < 4; ++mi) {
            int ar = m0 + mi * 16 + l16; if (ar >= M) ar = M - 1;
            const __bf16* ap = A + (size_t)ar * K + k0 + half * 16;
            const v8bf lo = *(const v8bf*)ap;
            const v8bf hi = *(const v8bf*)(ap + 8);
            v16bf a;
#pragma unroll
            for (int e = 0; e < 8; ++e) { a[e] = lo[e]; a[8 + e] = hi[e]; }
            afrag[mi] = a;
        }
        // B fragments (32x16 bf16): lane holds col n; K = half*16 .. +15 contig
        v16bf bfrag[2];
#pragma unroll
        for (int ni = 0; ni < 2; ++ni) {
            int col = n0 + ni * 16 + l16; if (col >= N) col = N - 1;
            const __bf16* bp = W + (size_t)col * K + k0 + half * 16;
            bfrag[ni] = *(const v16bf*)bp;
        }
        if (k0 + 32 < K) {  // global_prefetch_b8 for the next K step
            __builtin_prefetch(A + (size_t)(m0 + l16) * K + k0 + 32, 0, 3);
            __builtin_prefetch(W + (size_t)((n0 + l16) < N ? (n0 + l16) : (N - 1)) * K + k0 + 32, 0, 3);
        }
#pragma unroll
        for (int mi = 0; mi < 4; ++mi)
#pragma unroll
            for (int ni = 0; ni < 2; ++ni)
                acc[mi][ni] = __builtin_amdgcn_wmma_f32_16x16x32_bf16(
                    false, afrag[mi], false, bfrag[ni], (short)0, acc[mi][ni],
                    false, false);
    }

    // C/D layout: VGPR v, lane -> row = v + 8*half, col = l16
#pragma unroll
    for (int mi = 0; mi < 4; ++mi) {
#pragma unroll
        for (int ni = 0; ni < 2; ++ni) {
            const int col = n0 + ni * 16 + l16;
            if (col >= N) continue;
            const float bv = bias ? bias[col] : 0.f;
#pragma unroll
            for (int v = 0; v < 8; ++v) {
                const int row = m0 + mi * 16 + v + half * 8;
                if (row >= M) continue;
                float val = acc[mi][ni][v] + bv;
                if (addm) val += addm[(size_t)row * N + col];
                Out[(size_t)row * N + col] = val;
            }
        }
    }
}

// ----------------------- gates + state update + readout --------------------
__device__ __forceinline__ float soft_cap(float g) { return CAP * tanhf(g / CAP); }

__global__ __launch_bounds__(256)
void state_update_kernel(const float* __restrict__ fused,
                         const float* __restrict__ Cin,
                         const float* __restrict__ nin,
                         const float* __restrict__ min_,
                         float* __restrict__ Ct, float* __restrict__ nt,
                         float* __restrict__ mt, float* __restrict__ ht) {
    const int h   = blockIdx.x;
    const int b   = blockIdx.y;
    const int tid = threadIdx.x;

    __shared__ float sq[D], sk[D], sv[D], so[D], shnum[D];
    __shared__ float sc[4];  // ie, fe, m_t, h_den

    const float* frow = fused + (size_t)b * FOUT;
    const float rsD = rsqrtf((float)D);

    if (tid == 0) {
        float it = soft_cap(frow[h]);
        float ft = soft_cap(frow[H + h]);
        float mp = min_[b * H + h];
        float m_t = fmaxf(ft + mp, it);
        sc[0] = expf(it - m_t);
        sc[1] = expf(ft - m_t + mp);
        sc[2] = m_t;
        sc[3] = 0.f;
        mt[b * H + h] = m_t;
    }
    if (tid < D) {
        so[tid] = 1.f / (1.f + expf(-soft_cap(frow[2 * H + h * D + tid])));
        sq[tid] = frow[2 * H + HID + h * D + tid];
        sk[tid] = frow[2 * H + 2 * HID + h * D + tid] * rsD;
        sv[tid] = frow[2 * H + 3 * HID + h * D + tid];
        shnum[tid] = 0.f;
    }
    __syncthreads();

    const float ie = sc[0], fe = sc[1];

    if (tid < D) {
        const size_t nidx = ((size_t)b * H + h) * D + tid;
        float n_t = fe * nin[nidx] + ie * sk[tid];
        nt[nidx] = n_t;
        atomicAdd(&sc[3], n_t * sq[tid]);  // ds_add_f32
    }

    const size_t cbase = ((size_t)b * H + h) * (D * D);
    const float* Ci = Cin + cbase;
    float*       Co = Ct + cbase;
#pragma unroll 4
    for (int i = 0; i < (D * D) / 256; ++i) {  // 36 iters, 9216 elems
        const int idx = tid + i * 256;
        const int vv = idx / D, kk = idx - vv * D;
        float c = fe * Ci[idx] + ie * sv[vv] * sk[kk];
        Co[idx] = c;
        atomicAdd(&shnum[vv], c * sq[kk]);     // ds_add_f32 reduction
    }
    __syncthreads();

    if (tid < D) {
        float den = fmaxf(sc[3], 1.0f);
        ht[(size_t)b * HID + h * D + tid] = so[tid] * shnum[tid] / den;
    }
}

// ------------------------------- launcher ----------------------------------
extern "C" void kernel_launch(void* const* d_in, const int* in_sizes, int n_in,
                              void* d_out, int out_size, void* d_ws, size_t ws_size,
                              hipStream_t stream) {
    const float* x      = (const float*)d_in[0];
    const float* C      = (const float*)d_in[1];
    const float* n_st   = (const float*)d_in[2];
    const float* m_st   = (const float*)d_in[3];
    const float* ln1_w  = (const float*)d_in[4];
    const float* ln1_b  = (const float*)d_in[5];
    const float* W_up   = (const float*)d_in[6];
    const float* conv_w = (const float*)d_in[7];
    const float* W_skip = (const float*)d_in[8];
    const float* fW     = (const float*)d_in[9];
    const float* fb     = (const float*)d_in[10];
    const float* ln2_w  = (const float*)d_in[11];
    const float* ln2_b  = (const float*)d_in[12];
    const float* W_down = (const float*)d_in[13];

    const int bs = in_sizes[0] / INP;

    // outputs (concatenated flat: out, C_t, n_t, m_t)
    float* o_out = (float*)d_out;
    float* o_C   = o_out + (size_t)bs * INP;
    float* o_n   = o_C + (size_t)bs * H * D * D;
    float* o_m   = o_n + (size_t)bs * H * D;

    // workspace carve-up (256B-aligned slabs; ~47 MB total for bs=1024)
    char* p = (char*)d_ws;
    auto carve = [&](size_t bytes) -> void* {
        void* r = (void*)p;
        p += (bytes + 255) & ~(size_t)255;
        return r;
    };
    __bf16* Wup_bf   = (__bf16*)carve((size_t)PROJ * INP * 2);
    __bf16* Wskip_bf = (__bf16*)carve((size_t)HID * PROJ * 2);
    __bf16* fW_bf    = (__bf16*)carve((size_t)FOUT * PROJ * 2);
    __bf16* Wdown_bf = (__bf16*)carve((size_t)INP * HID * 2);
    __bf16* xn_bf    = (__bf16*)carve((size_t)bs * INP * 2);
    float*  xt       = (float*)carve((size_t)bs * PROJ * 4);
    __bf16* xc_bf    = (__bf16*)carve((size_t)bs * PROJ * 2);
    float*  fusedb   = (float*)carve((size_t)bs * FOUT * 4);
    float*  xskip    = (float*)carve((size_t)bs * HID * 4);
    float*  ht       = (float*)carve((size_t)bs * HID * 4);
    __bf16* hn_bf    = (__bf16*)carve((size_t)bs * HID * 2);
    (void)ws_size; (void)n_in; (void)out_size;

    // 1) weights -> bf16 (deterministic, every launch)
    {
        int n1 = PROJ * INP, n2 = HID * PROJ, n3 = FOUT * PROJ, n4 = INP * HID;
        f32_to_bf16_kernel<<<(n1 + 255) / 256, 256, 0, stream>>>(W_up, Wup_bf, n1);
        f32_to_bf16_kernel<<<(n2 + 255) / 256, 256, 0, stream>>>(W_skip, Wskip_bf, n2);
        f32_to_bf16_kernel<<<(n3 + 255) / 256, 256, 0, stream>>>(fW, fW_bf, n3);
        f32_to_bf16_kernel<<<(n4 + 255) / 256, 256, 0, stream>>>(W_down, Wdown_bf, n4);
    }

    // 2) LN1 -> bf16
    ln_bf16_kernel<<<bs, 256, 0, stream>>>(x, ln1_w, ln1_b, xn_bf, INP);

    // 3) x_t = x_n @ W_up^T   (bs x 1536 x 768)
    {
        dim3 g((PROJ + 127) / 128, (bs + 127) / 128);
        wmma_gemm_bf16_kernel<<<g, 256, 0, stream>>>(xn_bf, Wup_bf, xt,
                                                     nullptr, nullptr, bs, PROJ, INP);
    }

    // 4) causal conv + SiLU -> bf16
    {
        int tot = bs * PROJ;
        conv_silu_kernel<<<(tot + 255) / 256, 256, 0, stream>>>(xt, conv_w, xc_bf, tot);
    }

    // 5) fused = x_c @ fW^T + fb   (bs x 3088 x 1536)
    {
        dim3 g((FOUT + 127) / 128, (bs + 127) / 128);
        wmma_gemm_bf16_kernel<<<g, 256, 0, stream>>>(xc_bf, fW_bf, fusedb,
                                                     fb, nullptr, bs, FOUT, PROJ);
    }

    // 6) x_skip = x_c @ W_skip^T   (bs x 768 x 1536)
    {
        dim3 g((HID + 127) / 128, (bs + 127) / 128);
        wmma_gemm_bf16_kernel<<<g, 256, 0, stream>>>(xc_bf, Wskip_bf, xskip,
                                                     nullptr, nullptr, bs, HID, PROJ);
    }

    // 7) gates + C/n/m state update + h_t (the HBM-bound streaming pass)
    {
        dim3 g(H, bs);
        state_update_kernel<<<g, 256, 0, stream>>>(fusedb, C, n_st, m_st,
                                                   o_C, o_n, o_m, ht);
    }

    // 8) LN2 -> bf16
    ln_bf16_kernel<<<bs, 256, 0, stream>>>(ht, ln2_w, ln2_b, hn_bf, HID);

    // 9) out = h_n @ W_down^T + x_skip   (bs x 768 x 768)
    {
        dim3 g((INP + 127) / 128, (bs + 127) / 128);
        wmma_gemm_bf16_kernel<<<g, 256, 0, stream>>>(hn_bf, Wdown_bf, o_out,
                                                     nullptr, xskip, bs, INP, HID);
    }
}
